// TransformerBlock_1443109011789
// MI455X (gfx1250) — matrix operand order repaired and checked by reading, not compiled, not run on hardware
//
#include <hip/hip_runtime.h>
#include <cstdint>
#include <cstddef>

// ---------------------------------------------------------------------------
// Transformer block for MI455X (gfx1250, wave32, WMMA bf16, async LDS DMA)
// ---------------------------------------------------------------------------

typedef __attribute__((ext_vector_type(16))) __bf16 v16bf;
typedef __attribute__((ext_vector_type(8)))  __bf16 v8bf;
typedef __attribute__((ext_vector_type(8)))  float  v8f;

#define BATCH 128
#define CH    384
#define IHW   14
#define NTOK  196          // 14*14
#define HEADS 8
#define DH    32
#define INNER 256          // HEADS*DH
#define HID   1536         // 4*CH
#define MTOT  (BATCH * NTOK)   // 25088 token rows, divisible by 128
#define ATT_SCALE 0.17677669529663687f  // 32^-0.5

union U16 { v16bf v; v8bf h[2]; };

__device__ __forceinline__ v8f wmma_bf16(v16bf a, v16bf b, v8f c) {
  // D = A(16x32 bf16) x B(32x16 bf16) + C(16x16 f32)
  return __builtin_amdgcn_wmma_f32_16x16x32_bf16(
      /*neg_a=*/false, a, /*neg_b=*/false, b,
      /*c_mod=*/(short)0, c, /*reuse_a=*/false, /*reuse_b=*/false);
}

// Async global->LDS copy of 16B per lane (GLOBAL_LOAD_ASYNC_TO_LDS_B128,
// tracked by ASYNCcnt). lds_dst low 32 bits = wave-relative LDS offset.
__device__ __forceinline__ void async_ld128(void* lds_dst, const void* gsrc) {
  uint32_t l = (uint32_t)(uintptr_t)lds_dst;
  asm volatile("global_load_async_to_lds_b128 %0, %1, off"
               :: "v"(l), "v"(gsrc) : "memory");
}
__device__ __forceinline__ void wait_async0() {
  asm volatile("s_wait_asynccnt 0x0" ::: "memory");
}

// ---------------------------------------------------------------------------
// f32 -> bf16 conversion (transposing, for weights)
// ---------------------------------------------------------------------------
__global__ __launch_bounds__(256) void cvt_t_kernel(
    const float* __restrict__ in, __bf16* __restrict__ out, int K, int N) {
  // in: [K, N] f32   out: [N, K] bf16  (out[n*K + k] = in[k*N + n])
  int idx = blockIdx.x * 256 + threadIdx.x;
  if (idx < K * N) {
    int k = idx / N, n = idx - k * N;
    out[(size_t)n * K + k] = (__bf16)in[idx];
  }
}

// ---------------------------------------------------------------------------
// LayerNorm over C=384, one wave per token (12 channels per lane).
// chw_in=1: input is x laid out [B, C, n]; else token-major [M, C].
// ---------------------------------------------------------------------------
__global__ __launch_bounds__(256) void ln_kernel(
    const float* __restrict__ in, const float* __restrict__ w,
    const float* __restrict__ bias, __bf16* __restrict__ out, int chw_in) {
  int token = blockIdx.x * 8 + (threadIdx.x >> 5);
  int lane  = threadIdx.x & 31;
  int bb = token / NTOK, tt = token - bb * NTOK;
  float v[12];
  float s = 0.f, s2 = 0.f;
#pragma unroll
  for (int j = 0; j < 12; ++j) {
    int c = lane + 32 * j;
    float xv = chw_in ? in[((size_t)bb * CH + c) * NTOK + tt]
                      : in[(size_t)token * CH + c];
    v[j] = xv; s += xv; s2 += xv * xv;
  }
#pragma unroll
  for (int off = 16; off; off >>= 1) {
    s  += __shfl_xor(s,  off, 32);
    s2 += __shfl_xor(s2, off, 32);
  }
  float mu   = s * (1.f / CH);
  float var  = s2 * (1.f / CH) - mu * mu;
  float rstd = rsqrtf(var + 1e-6f);
#pragma unroll
  for (int j = 0; j < 12; ++j) {
    int c = lane + 32 * j;
    out[(size_t)token * CH + c] = (__bf16)((v[j] - mu) * rstd * w[c] + bias[c]);
  }
}

// ---------------------------------------------------------------------------
// Tiled bf16 WMMA GEMM: C[M,N] = A[M,K] (row-major) * B (given TRANSPOSED as
// Bt[N,K]). Both LDS tiles are plain row copies -> staged with async DMA,
// double buffered. 8 waves = 4(M) x 2(N); wave tile 32x64 = 2x4 WMMA tiles.
// Epilogues: 0=QKV scatter(bf16)  1=out-proj + residual(x CHW) -> xr f32
//            2=FFN1 bias+GELU(erf) -> bf16   3=FFN2 bias+residual -> CHW f32
// ---------------------------------------------------------------------------
template <int EPI>
__global__ __launch_bounds__(256) void gemm_kernel(
    const __bf16* __restrict__ A, const __bf16* __restrict__ Bt,
    int M, int N, int K,
    const float* __restrict__ bias, const float* __restrict__ resid,
    float* __restrict__ outf, __bf16* __restrict__ outb) {
  __shared__ __bf16 As[2][128][40];   // row stride 80B: 16B-aligned, bank-spread
  __shared__ __bf16 Bs[2][128][40];   // Bs[buf][n][k]

  const int tid  = threadIdx.x;
  const int wave = tid >> 5, lane = tid & 31;
  const int lm = lane & 15, lh = lane >> 4;
  const int wm = wave >> 1, wn = wave & 1;
  const int m0 = blockIdx.x * 128, n0 = blockIdx.y * 128;

  v8f acc[2][4] = {};

  const int row  = tid >> 1;            // 0..127 (tile row for both A and Bt)
  const int half = (tid & 1) * 16;      // 0 or 16 (bf16 elements)

  const __bf16* gA = A  + (size_t)(m0 + row) * K + half;
  const __bf16* gB = Bt + (size_t)(n0 + row) * K + half;

  auto stage = [&](int buf, int k0) {
    async_ld128(&As[buf][row][half],     gA + k0);
    async_ld128(&As[buf][row][half + 8], gA + k0 + 8);
    async_ld128(&Bs[buf][row][half],     gB + k0);
    async_ld128(&Bs[buf][row][half + 8], gB + k0 + 8);
  };

  stage(0, 0);
  wait_async0();
  __syncthreads();

  int cur = 0;
  for (int k0 = 0; k0 < K; k0 += 32) {
    const bool more = (k0 + 32 < K);
    if (more) stage(cur ^ 1, k0 + 32);   // overlap DMA with WMMA below

    v16bf af[2], bf[4];
#pragma unroll
    for (int mt = 0; mt < 2; ++mt) {
      int r = wm * 32 + mt * 16 + lm;
      U16 u;
      u.h[0] = *(const v8bf*)&As[cur][r][lh * 8];
      u.h[1] = *(const v8bf*)&As[cur][r][16 + lh * 8];
      af[mt] = u.v;
    }
#pragma unroll
    for (int nt = 0; nt < 4; ++nt) {
      int c = wn * 64 + nt * 16 + lm;
      U16 u;
      u.h[0] = *(const v8bf*)&Bs[cur][c][lh * 8];
      u.h[1] = *(const v8bf*)&Bs[cur][c][16 + lh * 8];
      bf[nt] = u.v;
    }
#pragma unroll
    for (int mt = 0; mt < 2; ++mt)
#pragma unroll
      for (int nt = 0; nt < 4; ++nt)
        acc[mt][nt] = wmma_bf16(af[mt], bf[nt], acc[mt][nt]);

    if (more) wait_async0();   // prefetched tile resident
    __syncthreads();           // all waves done reading 'cur' before reuse
    cur ^= 1;
  }

  // Epilogue. D layout: lane holds col (lane&15); VGPR r holds row 8*lh + r.
#pragma unroll
  for (int mt = 0; mt < 2; ++mt) {
#pragma unroll
    for (int nt = 0; nt < 4; ++nt) {
#pragma unroll
      for (int r = 0; r < 8; ++r) {
        int grow = m0 + wm * 32 + mt * 16 + 8 * lh + r;
        int gcol = n0 + wn * 64 + nt * 16 + lm;
        float val = acc[mt][nt][r];
        int bb = grow / NTOK, tt = grow - bb * NTOK;
        if (EPI == 0) {
          // scatter into [3][B][H][196][32] bf16 (q,k,v)
          int which = gcol >> 8, rem = gcol & 255;
          int hh = rem >> 5, dd = rem & 31;
          size_t idx =
              ((((size_t)which * BATCH + bb) * HEADS + hh) * NTOK + tt) * DH + dd;
          outb[idx] = (__bf16)val;
        } else if (EPI == 1) {
          val += bias[gcol] + resid[((size_t)bb * CH + gcol) * NTOK + tt];
          outf[(size_t)grow * CH + gcol] = val;     // xr, token-major
        } else if (EPI == 2) {
          val += bias[gcol];
          val = 0.5f * val * (1.f + erff(val * 0.70710678118654752f));
          outb[(size_t)grow * HID + gcol] = (__bf16)val;
        } else {
          val += bias[gcol] + resid[(size_t)grow * CH + gcol];
          outf[((size_t)bb * CH + gcol) * NTOK + tt] = val;   // d_out [B,C,H,W]
        }
      }
    }
  }
}

// ---------------------------------------------------------------------------
// Attention: one 128-thread (4-wave) block per (batch, head).
// scores = q @ k^T * scale + rel_bias ; softmax ; out = P @ v.
// ---------------------------------------------------------------------------
__global__ __launch_bounds__(128) void attn_kernel(
    const __bf16* __restrict__ Q, const __bf16* __restrict__ Kg,
    const __bf16* __restrict__ Vg, const float* __restrict__ btab,
    __bf16* __restrict__ outp) {
  __shared__ __bf16 Ks[208][40];        // K rows, 16B-aligned stride
  __shared__ __bf16 Vt[32][232];        // V transposed: Vt[d][token]
  __shared__ __bf16 Ps[4][16][232];     // per-wave softmax probabilities

  int bh = blockIdx.x;
  int b = bh >> 3, h = bh & 7;
  size_t base = (size_t)bh * NTOK * DH;
  int tid = threadIdx.x, wave = tid >> 5, lane = tid & 31;
  int lm = lane & 15, lh = lane >> 4;

  // zero-pad K rows 196..207 (plain DS stores), async-DMA the real rows
  for (int idx = tid; idx < 12 * 4; idx += 128) {
    int r = NTOK + (idx >> 2), p = (idx & 3) * 8;
    v8bf z = {};
    *(v8bf*)&Ks[r][p] = z;
  }
  for (int idx = tid; idx < NTOK * 4; idx += 128) {
    int r = idx >> 2, p = (idx & 3) * 8;
    async_ld128(&Ks[r][p], Kg + base + (size_t)r * DH + p);
  }
  // V transposed (register transpose on store)
  for (int idx = tid; idx < 224 * 4; idx += 128) {
    int r = idx >> 2, p = (idx & 3) * 8;
    v8bf val = {};
    if (r < NTOK) val = *(const v8bf*)(Vg + base + (size_t)r * DH + p);
#pragma unroll
    for (int i = 0; i < 8; ++i) Vt[p + i][r] = val[i];
  }
  wait_async0();
  __syncthreads();

  for (int it = wave; it < 13; it += 4) {
    // A fragment: q rows of this tile, straight from global
    v16bf aq = {};
    {
      int row = it * 16 + lm;
      if (row < NTOK) {
        const __bf16* qp = Q + base + (size_t)row * DH;
        U16 u;
        u.h[0] = *(const v8bf*)(qp + lh * 8);
        u.h[1] = *(const v8bf*)(qp + 16 + lh * 8);
        aq = u.v;
      }
    }
    // scores: 13 WMMA tiles (K = DH = 32 in one step)
    v8f sc[13];
#pragma unroll
    for (int j = 0; j < 13; ++j) {
      U16 u;
      u.h[0] = *(const v8bf*)&Ks[j * 16 + lm][lh * 8];
      u.h[1] = *(const v8bf*)&Ks[j * 16 + lm][16 + lh * 8];
      v8f z = {};
      sc[j] = wmma_bf16(aq, u.v, z);
    }
    // scale + relative bias + mask, then rowwise softmax
    float rsum[8];
    int gi0 = it * 16 + 8 * lh;
#pragma unroll
    for (int r = 0; r < 8; ++r) {
      int gi = gi0 + r;
      int yi = gi / IHW, xi = gi - yi * IHW;
      float mx = -1e30f;
#pragma unroll
      for (int j = 0; j < 13; ++j) {
        int gj = j * 16 + lm;
        float val = -1e30f;
        if (gi < NTOK && gj < NTOK) {
          int yj = gj / IHW, xj = gj - yj * IHW;
          int ridx = (yi - yj + IHW - 1) * (2 * IHW - 1) + (xi - xj + IHW - 1);
          val = sc[j][r] * ATT_SCALE + btab[ridx * HEADS + h];
        }
        sc[j][r] = val;
        mx = fmaxf(mx, val);
      }
#pragma unroll
      for (int m = 1; m <= 8; m <<= 1) mx = fmaxf(mx, __shfl_xor(mx, m, 32));
      float sum = 0.f;
#pragma unroll
      for (int j = 0; j < 13; ++j) {
        float p = __expf(sc[j][r] - mx);
        sc[j][r] = p;
        sum += p;
      }
#pragma unroll
      for (int m = 1; m <= 8; m <<= 1) sum += __shfl_xor(sum, m, 32);
      rsum[r] = sum;
    }
    // restage probabilities as bf16 A-operand tiles (per-wave LDS scratch)
#pragma unroll
    for (int j = 0; j < 13; ++j)
#pragma unroll
      for (int r = 0; r < 8; ++r)
        Ps[wave][8 * lh + r][j * 16 + lm] = (__bf16)sc[j][r];
#pragma unroll
    for (int r = 0; r < 8; ++r)
      Ps[wave][8 * lh + r][208 + lm] = (__bf16)0.f;   // zero pad 208..223

    // out = P @ V : K = 224 = 7 WMMA chunks, two 16-col d tiles
    v8f o0 = {}, o1 = {};
#pragma unroll
    for (int kc = 0; kc < 7; ++kc) {
      U16 ua, ub0, ub1;
      ua.h[0]  = *(const v8bf*)&Ps[wave][lm][kc * 32 + lh * 8];
      ua.h[1]  = *(const v8bf*)&Ps[wave][lm][kc * 32 + 16 + lh * 8];
      ub0.h[0] = *(const v8bf*)&Vt[lm][kc * 32 + lh * 8];
      ub0.h[1] = *(const v8bf*)&Vt[lm][kc * 32 + 16 + lh * 8];
      ub1.h[0] = *(const v8bf*)&Vt[16 + lm][kc * 32 + lh * 8];
      ub1.h[1] = *(const v8bf*)&Vt[16 + lm][kc * 32 + 16 + lh * 8];
      o0 = wmma_bf16(ua.v, ub0.v, o0);
      o1 = wmma_bf16(ua.v, ub1.v, o1);
    }
#pragma unroll
    for (int r = 0; r < 8; ++r) {
      int gi = gi0 + r;
      if (gi < NTOK) {
        float inv = 1.f / rsum[r];
        size_t o = ((size_t)b * NTOK + gi) * INNER + h * DH;
        outp[o + lm]      = (__bf16)(o0[r] * inv);
        outp[o + 16 + lm] = (__bf16)(o1[r] * inv);
      }
    }
  }
}

// ---------------------------------------------------------------------------
// Launcher
// ---------------------------------------------------------------------------
extern "C" void kernel_launch(void* const* d_in, const int* in_sizes, int n_in,
                              void* d_out, int out_size, void* d_ws,
                              size_t ws_size, hipStream_t stream) {
  (void)in_sizes; (void)n_in; (void)out_size; (void)ws_size;
  const float* x    = (const float*)d_in[0];
  const float* ln1w = (const float*)d_in[1];
  const float* ln1b = (const float*)d_in[2];
  const float* wqkv = (const float*)d_in[3];
  const float* btab = (const float*)d_in[4];
  const float* wout = (const float*)d_in[5];
  const float* bout = (const float*)d_in[6];
  const float* ln2w = (const float*)d_in[7];
  const float* ln2b = (const float*)d_in[8];
  const float* w1   = (const float*)d_in[9];
  const float* b1   = (const float*)d_in[10];
  const float* w2   = (const float*)d_in[11];
  const float* b2   = (const float*)d_in[12];

  char* ws = (char*)d_ws;
  size_t off = 0;
  auto carve = [&](size_t bytes) -> char* {
    char* p = ws + off;
    off += bytes;
    off = (off + 255) & ~(size_t)255;
    return p;
  };

  __bf16* wqkv_t  = (__bf16*)carve((size_t)CH * 768 * 2);     // [768][384]
  __bf16* wout_t  = (__bf16*)carve((size_t)INNER * CH * 2);   // [384][256]
  __bf16* w1_t    = (__bf16*)carve((size_t)CH * HID * 2);     // [1536][384]
  __bf16* w2_t    = (__bf16*)carve((size_t)HID * CH * 2);     // [384][1536]
  __bf16* ybuf    = (__bf16*)carve((size_t)MTOT * CH * 2);    // LN1/LN2 out
  __bf16* qkvbuf  = (__bf16*)carve((size_t)3 * BATCH * HEADS * NTOK * DH * 2);
  __bf16* attnout = (__bf16*)carve((size_t)MTOT * INNER * 2);
  float*  xr      = (float*) carve((size_t)MTOT * CH * 4);    // post-attn residual
  __bf16* h1      = (__bf16*)carve((size_t)MTOT * HID * 2);

  // weights converted to bf16 AND transposed to [N][K]
  cvt_t_kernel<<<(CH * 768 + 255) / 256, 256, 0, stream>>>(wqkv, wqkv_t, CH, 768);
  cvt_t_kernel<<<(INNER * CH + 255) / 256, 256, 0, stream>>>(wout, wout_t, INNER, CH);
  cvt_t_kernel<<<(CH * HID + 255) / 256, 256, 0, stream>>>(w1, w1_t, CH, HID);
  cvt_t_kernel<<<(HID * CH + 255) / 256, 256, 0, stream>>>(w2, w2_t, HID, CH);

  // LN1: x [B,C,n] -> ybuf bf16 [M,C]
  ln_kernel<<<MTOT / 8, 256, 0, stream>>>(x, ln1w, ln1b, ybuf, 1);

  // QKV: [M,384] x [384,768] -> scatter q/k/v
  gemm_kernel<0><<<dim3(MTOT / 128, 768 / 128), 256, 0, stream>>>(
      ybuf, wqkv_t, MTOT, 768, CH, nullptr, nullptr, nullptr, qkvbuf);

  const __bf16* Qb = qkvbuf;
  const __bf16* Kb = qkvbuf + (size_t)BATCH * HEADS * NTOK * DH;
  const __bf16* Vb = Kb + (size_t)BATCH * HEADS * NTOK * DH;
  attn_kernel<<<BATCH * HEADS, 128, 0, stream>>>(Qb, Kb, Vb, btab, attnout);

  // out-proj + residual with x -> xr (token-major f32)
  gemm_kernel<1><<<dim3(MTOT / 128, CH / 128), 256, 0, stream>>>(
      attnout, wout_t, MTOT, CH, INNER, bout, x, xr, nullptr);

  // LN2: xr [M,C] -> ybuf bf16
  ln_kernel<<<MTOT / 8, 256, 0, stream>>>(xr, ln2w, ln2b, ybuf, 0);

  // FFN1 + GELU -> h1 bf16
  gemm_kernel<2><<<dim3(MTOT / 128, HID / 128), 256, 0, stream>>>(
      ybuf, w1_t, MTOT, HID, CH, b1, nullptr, nullptr, h1);

  // FFN2 + residual -> d_out [B,C,H,W] f32
  gemm_kernel<3><<<dim3(MTOT / 128, CH / 128), 256, 0, stream>>>(
      h1, w2_t, MTOT, CH, HID, b2, xr, (float*)d_out, nullptr);
}
